// SelfAttention_49349174231647
// MI455X (gfx1250) — compile-verified, matrix operand order checked by reading, not run on hardware
//
#include <hip/hip_runtime.h>
#include <hip/hip_bf16.h>

// ---------------------------------------------------------------------------
// Causal multi-head self-attention forward for MI455X (gfx1250, wave32).
// - All matmuls via v_wmma_f32_16x16x32_bf16 (fp32 accumulate).
// - GEMMs: 32x64 wave tiles, register double-buffered k-loop.
// - Attention: block-per-(b,h,128 q rows); K/V tiles staged into LDS by the
//   Tensor Data Mover (tensor_load_to_lds + s_wait_tensorcnt), double-buffered.
// ---------------------------------------------------------------------------

typedef __attribute__((ext_vector_type(16))) __bf16 v16bf;
typedef __attribute__((ext_vector_type(8)))  float  v8f;
typedef __attribute__((ext_vector_type(4)))  unsigned int u32x4;
typedef __attribute__((ext_vector_type(8)))  int i32x8;
typedef __attribute__((ext_vector_type(4)))  int i32x4;

constexpr int Bc = 2;
constexpr int Tc = 2048;
constexpr int Ec = 1024;
constexpr int Hc = 16;
constexpr int HSc = 64;          // head size
constexpr int BTc = Bc * Tc;     // 4096 rows

union Frag {
    v16bf  v;
    uint4  q[2];
    __bf16 e[16];
};

// --------------------------- fp32 -> bf16 conversion -----------------------
__global__ void cvt_bf16_kernel(const float* __restrict__ src,
                                __bf16* __restrict__ dst, int n) {
    int i = blockIdx.x * blockDim.x + threadIdx.x;
    if (i < n) dst[i] = (__bf16)src[i];
}

// Weight transpose + convert: Wt[n][k] = W[k][n]  (B-fragments then load
// contiguously along K, matching the 16-bit WMMA B VGPR layout).
__global__ void cvt_wT_kernel(const float* __restrict__ W,
                              __bf16* __restrict__ Wt) {
    int i = blockIdx.x * blockDim.x + threadIdx.x;   // over E*E
    int k = i >> 10;
    int n = i & (Ec - 1);
    Wt[n * Ec + k] = (__bf16)W[k * Ec + n];
}

// --------------------------- TDM 2D tile load ------------------------------
// Builds a 2D Tensor DMA Descriptor (D#) per CDNA5 ISA
// and issues TENSOR_LOAD_TO_LDS. data_size = 2 bytes (bf16).
// This toolchain exposes the 6-arg builtin:
//   (u32x4 group0, i32x8 group1, i32x4 group2, i32x4 group3, i32x8 extra, i32 cpol)
__device__ __forceinline__ void tdm_load_2d(unsigned lds_off, const void* gaddr,
                                            unsigned dim0, unsigned dim1,
                                            unsigned tile0, unsigned tile1,
                                            unsigned stride0) {
    unsigned long long ga = (unsigned long long)(uintptr_t)gaddr;
    u32x4 g0;
    g0[0] = 1u;                                         // count=1, user D#
    g0[1] = lds_off;                                    // lds_addr (bytes)
    g0[2] = (unsigned)(ga & 0xFFFFFFFFu);               // global_addr[31:0]
    g0[3] = (unsigned)((ga >> 32) & 0x1FFFFFFu)         // global_addr[56:32]
          | (2u << 30);                                 // type = 2 ("image")
    i32x8 g1;
    g1[0] = (int)(1u << 16);                            // data_size=1 -> 2B
    g1[1] = (int)((dim0 & 0xFFFFu) << 16);              // tensor_dim0[15:0]
    g1[2] = (int)(((dim0 >> 16) & 0xFFFFu)              // tensor_dim0[31:16]
          | ((dim1 & 0xFFFFu) << 16));                  // tensor_dim1[15:0]
    g1[3] = (int)(((dim1 >> 16) & 0xFFFFu)              // tensor_dim1[31:16]
          | ((tile0 & 0xFFFFu) << 16));                 // tile_dim0
    g1[4] = (int)(tile1 & 0xFFFFu);                     // tile_dim1 (tile_dim2=0)
    g1[5] = (int)stride0;                               // tensor_dim0_stride[31:0]
    g1[6] = 0;                                          // stride0[47:32], dim1_stride lo
    g1[7] = 0;                                          // dim1_stride hi
    i32x4 z4 = {0, 0, 0, 0};
    i32x8 z8 = {0, 0, 0, 0, 0, 0, 0, 0};
    __builtin_amdgcn_tensor_load_to_lds(g0, g1, z4, z4, z8, 0);
}

// --------------------------- fused QKV projection --------------------------
// One wave computes a 32x64 tile of X @ Wt; k-loop register double-buffered.
// Q,K written [b][h][t][d]; V written transposed [b][h][d][t].
__global__ void __launch_bounds__(256) gemm_qkv_kernel(
    const __bf16* __restrict__ Xq, const __bf16* __restrict__ Xk,
    const __bf16* __restrict__ Xv,
    const __bf16* __restrict__ Wq, const __bf16* __restrict__ Wk,
    const __bf16* __restrict__ Wv,
    __bf16* __restrict__ Q, __bf16* __restrict__ K, __bf16* __restrict__ Vt) {
    const int mode = blockIdx.y;                 // 0=Q 1=K 2=V
    const __bf16* X = (mode == 0) ? Xq : (mode == 1) ? Xk : Xv;
    const __bf16* W = (mode == 0) ? Wq : (mode == 1) ? Wk : Wv;

    int wave = (blockIdx.x * blockDim.x + threadIdx.x) >> 5;   // 0..2047
    int lane = threadIdx.x & 31;
    int lrow = lane & 15;
    int lhi  = lane >> 4;
    int m0 = (wave >> 4) << 5;    // 128 m-tiles of 32 rows
    int n0 = (wave & 15) << 6;    // 16 groups of 64 output cols

    const __bf16* arow = X + (size_t)(m0 + lrow) * Ec + lhi * 8;
    const __bf16* brow = W + (size_t)(n0 + lrow) * Ec + lhi * 8;

    v8f acc[2][4] = {};
    Frag a[2][2], b[2][4];
#pragma unroll
    for (int i = 0; i < 2; ++i) {
        const __bf16* p = arow + i * 16 * Ec;
        a[0][i].q[0] = *(const uint4*)p;
        a[0][i].q[1] = *(const uint4*)(p + 16);
    }
#pragma unroll
    for (int j = 0; j < 4; ++j) {
        const __bf16* p = brow + j * 16 * Ec;
        b[0][j].q[0] = *(const uint4*)p;
        b[0][j].q[1] = *(const uint4*)(p + 16);
    }
    for (int k0 = 0; k0 < Ec; k0 += 32) {
        int cur = (k0 >> 5) & 1, nxt = cur ^ 1;
        if (k0 + 32 < Ec) {       // prefetch next k-step into the other buffer
#pragma unroll
            for (int i = 0; i < 2; ++i) {
                const __bf16* p = arow + i * 16 * Ec + (k0 + 32);
                a[nxt][i].q[0] = *(const uint4*)p;
                a[nxt][i].q[1] = *(const uint4*)(p + 16);
            }
#pragma unroll
            for (int j = 0; j < 4; ++j) {
                const __bf16* p = brow + j * 16 * Ec + (k0 + 32);
                b[nxt][j].q[0] = *(const uint4*)p;
                b[nxt][j].q[1] = *(const uint4*)(p + 16);
            }
        }
#pragma unroll
        for (int i = 0; i < 2; ++i)
#pragma unroll
            for (int j = 0; j < 4; ++j)
                acc[i][j] = __builtin_amdgcn_wmma_f32_16x16x32_bf16(
                    false, a[cur][i].v, false, b[cur][j].v, (short)0,
                    acc[i][j], false, false);
    }
#pragma unroll
    for (int i = 0; i < 2; ++i)
#pragma unroll
        for (int j = 0; j < 4; ++j)
#pragma unroll
            for (int r = 0; r < 8; ++r) {
                int M = m0 + i * 16 + r + lhi * 8;   // global token row
                int N = n0 + j * 16 + lrow;          // output feature
                int bb = M >> 11, t = M & (Tc - 1);
                int h = N >> 6, d = N & 63;
                __bf16 val = (__bf16)acc[i][j][r];
                if (mode == 2) {
                    Vt[(((size_t)(bb * Hc + h) * HSc + d) << 11) + t] = val;
                } else {
                    __bf16* dst = (mode == 0) ? Q : K;
                    dst[(((size_t)(bb * Hc + h) << 11) + t) * HSc + d] = val;
                }
            }
}

// --------------------------- flash attention -------------------------------
// Block = one (b,h) and 128 consecutive q rows (8 waves x 16-row q-tiles).
// K/V tiles for each 32-key chunk are DMA'd into LDS by the TDM (wave 0),
// double-buffered one chunk ahead; all 8 waves consume the shared stage.
__global__ void __launch_bounds__(256) attn_kernel(
    const __bf16* __restrict__ Q, const __bf16* __restrict__ K,
    const __bf16* __restrict__ Vt, __bf16* __restrict__ Y) {
    // LDS: K stage 2x4KB | V stage 2x4KB | P transpose 8x1KB = 24KB
    __shared__ alignas(16) char smem[24576];

    int wave = threadIdx.x >> 5;
    int lane = threadIdx.x & 31;
    int lrow = lane & 15;
    int lhi  = lane >> 4;
    int bh = blockIdx.x >> 4;          // 0..31  (b*H + h)
    int qg = blockIdx.x & 15;          // 16 q-groups of 128 rows
    int q0 = qg * 128 + wave * 16;

    const __bf16* Qb = Q + (size_t)bh * Tc * HSc;
    const __bf16* Kb = K + (size_t)bh * Tc * HSc;
    const __bf16* Vb = Vt + (size_t)bh * HSc * Tc;
    __bf16* plds = (__bf16*)(smem + 16384 + wave * 1024);

    // Q A-fragments: two K-halves of this wave's 16x64 q-tile (loaded once).
    Frag aq[2];
#pragma unroll
    for (int g = 0; g < 2; ++g) {
        const __bf16* p = Qb + (size_t)(q0 + lrow) * HSc + g * 32 + lhi * 8;
        aq[g].q[0] = *(const uint4*)p;
        aq[g].q[1] = *(const uint4*)(p + 16);
    }

    v8f m, l;
    v8f o[4] = {};
#pragma unroll
    for (int r = 0; r < 8; ++r) { m[r] = -3.0e38f; l[r] = 0.0f; }

    const int nchunks = qg * 4 + 4;    // covers keys [0, qg*128+128)

    // Prologue: TDM-load chunk 0 into buffer 0 (wave 0 only; 2 descriptors).
    if (wave == 0) {
        tdm_load_2d(0u, Kb, HSc, Tc, HSc, 32, HSc);                // K: 32x64
        tdm_load_2d(8192u, Vb, Tc, HSc, 32, HSc, Tc);              // V: 64x32
    }

    for (int c = 0; c < nchunks; ++c) {
        int k0 = c * 32;
        int cur = c & 1, nxt = cur ^ 1;
        if (wave == 0) {
            if (c + 1 < nchunks) {     // prefetch next chunk into other buffer
                int kn = k0 + 32;
                tdm_load_2d((unsigned)(nxt * 4096), Kb + (size_t)kn * HSc,
                            HSc, Tc - kn, HSc, 32, HSc);
                tdm_load_2d((unsigned)(8192 + nxt * 4096), Vb + kn,
                            Tc - kn, HSc, 32, HSc, Tc);
                __builtin_amdgcn_s_wait_tensorcnt(2);  // chunk c finished
            } else {
                __builtin_amdgcn_s_wait_tensorcnt(0);
            }
        }
        __syncthreads();               // stage for chunk c visible to all waves

        const __bf16* ks = (const __bf16*)(smem + cur * 4096);         // [key][d]
        const __bf16* vs = (const __bf16*)(smem + 8192 + cur * 4096);  // [d][key]

        // ---- S = Q K^T for two 16-key sub-tiles (4 WMMAs, B from LDS) ----
        v8f s[2];
#pragma unroll
        for (int hB = 0; hB < 2; ++hB) {
            Frag bk0, bk1;
            const __bf16* kp = ks + (hB * 16 + lrow) * HSc + lhi * 8;
            bk0.q[0] = *(const uint4*)kp;        bk0.q[1] = *(const uint4*)(kp + 16);
            bk1.q[0] = *(const uint4*)(kp + 32); bk1.q[1] = *(const uint4*)(kp + 48);
            v8f z = {};
            v8f t0 = __builtin_amdgcn_wmma_f32_16x16x32_bf16(
                false, aq[0].v, false, bk0.v, (short)0, z, false, false);
            s[hB] = __builtin_amdgcn_wmma_f32_16x16x32_bf16(
                false, aq[1].v, false, bk1.v, (short)0, t0, false, false);
        }
        // ---- scale + causal mask ----
#pragma unroll
        for (int hB = 0; hB < 2; ++hB) {
            int key = k0 + hB * 16 + lrow;
#pragma unroll
            for (int r = 0; r < 8; ++r) {
                int qrow = q0 + r + lhi * 8;
                float v = s[hB][r] * 0.125f;           // 1/sqrt(64)
                s[hB][r] = (key > qrow) ? -1.0e10f : v;
            }
        }
        // ---- online softmax (row reductions across 16 column lanes) ----
        v8f mc;
#pragma unroll
        for (int r = 0; r < 8; ++r) {
            float v = fmaxf(s[0][r], s[1][r]);
            v = fmaxf(v, __shfl_xor(v, 1, 32));
            v = fmaxf(v, __shfl_xor(v, 2, 32));
            v = fmaxf(v, __shfl_xor(v, 4, 32));
            v = fmaxf(v, __shfl_xor(v, 8, 32));
            mc[r] = v;
        }
        v8f f;
#pragma unroll
        for (int r = 0; r < 8; ++r) {
            float mn = fmaxf(m[r], mc[r]);
            f[r] = __expf(m[r] - mn);
            m[r] = mn;
        }
#pragma unroll
        for (int r = 0; r < 8; ++r) {
            float p0 = __expf(s[0][r] - m[r]);
            float p1 = __expf(s[1][r] - m[r]);
            s[0][r] = p0; s[1][r] = p1;
            float v = p0 + p1;
            v += __shfl_xor(v, 1, 32);
            v += __shfl_xor(v, 2, 32);
            v += __shfl_xor(v, 4, 32);
            v += __shfl_xor(v, 8, 32);
            l[r] = l[r] * f[r] + v;
        }
#pragma unroll
        for (int j = 0; j < 4; ++j)
#pragma unroll
            for (int r = 0; r < 8; ++r) o[j][r] *= f[r];

        // ---- transpose P (D-layout) -> A-layout via per-wave LDS ----
#pragma unroll
        for (int r = 0; r < 8; ++r) {
            int M = r + lhi * 8;
            plds[M * 32 + lrow]      = (__bf16)s[0][r];
            plds[M * 32 + 16 + lrow] = (__bf16)s[1][r];
        }
        asm volatile("s_wait_dscnt 0" ::: "memory");
        Frag pa;
        {
            const __bf16* p = plds + lrow * 32 + lhi * 8;
            pa.q[0] = *(const uint4*)p;
            pa.q[1] = *(const uint4*)(p + 16);
        }
        // ---- O += P @ V (B-fragments from the shared V stage) ----
#pragma unroll
        for (int j = 0; j < 4; ++j) {
            Frag bv;
            const __bf16* vp = vs + (j * 16 + lrow) * 32 + lhi * 8;
            bv.q[0] = *(const uint4*)vp;
            bv.q[1] = *(const uint4*)(vp + 16);
            o[j] = __builtin_amdgcn_wmma_f32_16x16x32_bf16(
                false, pa.v, false, bv.v, (short)0, o[j], false, false);
        }
        __syncthreads();   // everyone done with buffer `cur` before TDM reuse
    }

    // ---- epilogue: normalize and store bf16 Y[b][t][h*64+d] ----
    v8f inv;
#pragma unroll
    for (int r = 0; r < 8; ++r) inv[r] = 1.0f / l[r];
    int b = bh >> 4;
    int h = bh & 15;
#pragma unroll
    for (int j = 0; j < 4; ++j)
#pragma unroll
        for (int r = 0; r < 8; ++r) {
            int t = q0 + r + lhi * 8;
            int d = j * 16 + lrow;
            Y[((size_t)(b * Tc + t)) * Ec + h * HSc + d] = (__bf16)(o[j][r] * inv[r]);
        }
}

// --------------------------- output projection -----------------------------
__global__ void __launch_bounds__(256) gemm_out_kernel(
    const __bf16* __restrict__ Ybf, const __bf16* __restrict__ Wp,
    float* __restrict__ out) {
    int wave = (blockIdx.x * blockDim.x + threadIdx.x) >> 5;   // 0..2047
    int lane = threadIdx.x & 31;
    int lrow = lane & 15;
    int lhi  = lane >> 4;
    int m0 = (wave >> 4) << 5;
    int n0 = (wave & 15) << 6;

    const __bf16* arow = Ybf + (size_t)(m0 + lrow) * Ec + lhi * 8;
    const __bf16* brow = Wp + (size_t)(n0 + lrow) * Ec + lhi * 8;

    v8f acc[2][4] = {};
    Frag a[2][2], b[2][4];
#pragma unroll
    for (int i = 0; i < 2; ++i) {
        const __bf16* p = arow + i * 16 * Ec;
        a[0][i].q[0] = *(const uint4*)p;
        a[0][i].q[1] = *(const uint4*)(p + 16);
    }
#pragma unroll
    for (int j = 0; j < 4; ++j) {
        const __bf16* p = brow + j * 16 * Ec;
        b[0][j].q[0] = *(const uint4*)p;
        b[0][j].q[1] = *(const uint4*)(p + 16);
    }
    for (int k0 = 0; k0 < Ec; k0 += 32) {
        int cur = (k0 >> 5) & 1, nxt = cur ^ 1;
        if (k0 + 32 < Ec) {
#pragma unroll
            for (int i = 0; i < 2; ++i) {
                const __bf16* p = arow + i * 16 * Ec + (k0 + 32);
                a[nxt][i].q[0] = *(const uint4*)p;
                a[nxt][i].q[1] = *(const uint4*)(p + 16);
            }
#pragma unroll
            for (int j = 0; j < 4; ++j) {
                const __bf16* p = brow + j * 16 * Ec + (k0 + 32);
                b[nxt][j].q[0] = *(const uint4*)p;
                b[nxt][j].q[1] = *(const uint4*)(p + 16);
            }
        }
#pragma unroll
        for (int i = 0; i < 2; ++i)
#pragma unroll
            for (int j = 0; j < 4; ++j)
                acc[i][j] = __builtin_amdgcn_wmma_f32_16x16x32_bf16(
                    false, a[cur][i].v, false, b[cur][j].v, (short)0,
                    acc[i][j], false, false);
    }
#pragma unroll
    for (int i = 0; i < 2; ++i)
#pragma unroll
        for (int j = 0; j < 4; ++j)
#pragma unroll
            for (int r = 0; r < 8; ++r) {
                int M = m0 + i * 16 + r + lhi * 8;
                int N = n0 + j * 16 + lrow;
                out[(size_t)M * Ec + N] = acc[i][j][r];
            }
}

// ---------------------------------------------------------------------------
extern "C" void kernel_launch(void* const* d_in, const int* in_sizes, int n_in,
                              void* d_out, int out_size, void* d_ws, size_t ws_size,
                              hipStream_t stream) {
    (void)in_sizes; (void)n_in; (void)out_size; (void)ws_size;

    const float* q_in = (const float*)d_in[0];
    const float* k_in = (const float*)d_in[1];
    const float* v_in = (const float*)d_in[2];
    const float* Wq   = (const float*)d_in[3];
    const float* Wk   = (const float*)d_in[4];
    const float* Wv   = (const float*)d_in[5];
    const float* Wp   = (const float*)d_in[6];
    float* out = (float*)d_out;

    // Workspace layout (bf16 scratch), ~64 MB total.
    char* ws = (char*)d_ws;
    const size_t NX = (size_t)BTc * Ec;
    const size_t NW = (size_t)Ec * Ec;
    const size_t NQ = (size_t)Bc * Hc * Tc * HSc;
    size_t off = 0;
    __bf16* Xq  = (__bf16*)(ws + off); off += NX * 2;
    __bf16* Xk  = (__bf16*)(ws + off); off += NX * 2;
    __bf16* Xv  = (__bf16*)(ws + off); off += NX * 2;
    __bf16* Wtq = (__bf16*)(ws + off); off += NW * 2;
    __bf16* Wtk = (__bf16*)(ws + off); off += NW * 2;
    __bf16* Wtv = (__bf16*)(ws + off); off += NW * 2;
    __bf16* Wtp = (__bf16*)(ws + off); off += NW * 2;
    __bf16* Qb  = (__bf16*)(ws + off); off += NQ * 2;
    __bf16* Kb  = (__bf16*)(ws + off); off += NQ * 2;
    __bf16* Vtb = (__bf16*)(ws + off); off += NQ * 2;
    __bf16* Yb  = (__bf16*)(ws + off); off += NX * 2;

    cvt_bf16_kernel<<<(int)(NX / 256), 256, 0, stream>>>(q_in, Xq, (int)NX);
    cvt_bf16_kernel<<<(int)(NX / 256), 256, 0, stream>>>(k_in, Xk, (int)NX);
    cvt_bf16_kernel<<<(int)(NX / 256), 256, 0, stream>>>(v_in, Xv, (int)NX);
    cvt_wT_kernel<<<(int)(NW / 256), 256, 0, stream>>>(Wq, Wtq);
    cvt_wT_kernel<<<(int)(NW / 256), 256, 0, stream>>>(Wk, Wtk);
    cvt_wT_kernel<<<(int)(NW / 256), 256, 0, stream>>>(Wv, Wtv);
    cvt_wT_kernel<<<(int)(NW / 256), 256, 0, stream>>>(Wp, Wtp);

    // QKV projection: 2048 waves per matrix -> 256 blocks, grid.y = Q/K/V.
    gemm_qkv_kernel<<<dim3(256, 3), 256, 0, stream>>>(Xq, Xk, Xv, Wtq, Wtk, Wtv,
                                                      Qb, Kb, Vtb);
    // Attention: 32 (b,h) x 16 q-groups = 512 blocks of 8 waves.
    attn_kernel<<<512, 256, 0, stream>>>(Qb, Kb, Vtb, Yb);
    // Output projection -> fp32 d_out.
    gemm_out_kernel<<<256, 256, 0, stream>>>(Yb, Wtp, out);
}
